// ExpertGating_37864431681940
// MI455X (gfx1250) — matrix operand order repaired and checked by reading, not compiled, not run on hardware
//
#include <hip/hip_runtime.h>

// ---------------------------------------------------------------------------
// MoE expert gating for MI455X (gfx1250, wave32, WMMA).
// Pipeline:
//   1) cvt hidden_states f32 -> bf16 row-major           (ws: Abf)
//   2) transpose+cvt W1 -> W1T bf16 [n][k]               (ws: W1T)
//   3) build zero-padded W2T bf16 [16][1024]             (ws: W2T)
//   4) router: fused GEMM1(relu,bias) + GEMM2 + softmax + top2, all WMMA bf16
//   5) combine: out = g0*EO[i0] + g1*EO[i1]  (float4 gather)
// ---------------------------------------------------------------------------

#define NTOK 8192   // B*S = 4*2048
#define HD   1024   // hidden dim
#define NEXP 8      // experts

typedef __attribute__((ext_vector_type(16))) __bf16 v16bf;
typedef __attribute__((ext_vector_type(8)))  float  v8f;

union ABFrag { v16bf v; uint4 q[2]; };

__device__ __forceinline__ unsigned short f2bf(float f) {
  unsigned u = __float_as_uint(f);
  return (unsigned short)((u + 0x7FFFu + ((u >> 16) & 1u)) >> 16); // RNE
}

// --------------------------- preconvert kernels ----------------------------

__global__ void __launch_bounds__(256)
cvt_bf16_kernel(const float* __restrict__ in, unsigned short* __restrict__ out, int n) {
  int i = (blockIdx.x * 256 + threadIdx.x) * 4;
  if (i + 3 < n) {
    float4 v = *(const float4*)(in + i);
    uint2 p;
    p.x = (unsigned)f2bf(v.x) | ((unsigned)f2bf(v.y) << 16);
    p.y = (unsigned)f2bf(v.z) | ((unsigned)f2bf(v.w) << 16);
    *(uint2*)(out + i) = p;
  }
}

__global__ void __launch_bounds__(256)
transpose_w1_kernel(const float* __restrict__ W1, unsigned short* __restrict__ W1T) {
  int idx = blockIdx.x * 256 + threadIdx.x;   // idx = n*HD + k
  int n = idx >> 10, k = idx & (HD - 1);
  W1T[idx] = f2bf(W1[k * HD + n]);
}

__global__ void __launch_bounds__(256)
prep_w2_kernel(const float* __restrict__ W2, unsigned short* __restrict__ W2T) {
  int idx = blockIdx.x * 256 + threadIdx.x;   // idx = e*HD + k, e in [0,16)
  int e = idx >> 10, k = idx & (HD - 1);
  W2T[idx] = (e < NEXP) ? f2bf(W2[k * NEXP + e]) : (unsigned short)0;
}

// ------------------------------ router kernel ------------------------------
// 256 threads = 8 waves; each wave owns 16 tokens. Per 64-col group:
//   GEMM1: 4 accumulators x 32 K-steps of v_wmma_f32_16x16x32_bf16
//   bias + relu, bf16 round-trip through LDS (D-layout -> A-layout)
//   GEMM2: 2 x wmma against padded W2T accumulating the logits tile
// Then per-token softmax + top-2 on lanes 0..15.

__global__ void __launch_bounds__(256)
router_kernel(const unsigned short* __restrict__ Abf,   // [NTOK][HD] bf16
              const unsigned short* __restrict__ W1T,   // [HD(n)][HD(k)] bf16
              const unsigned short* __restrict__ W2T,   // [16(e)][HD(k)] bf16
              const float* __restrict__ b1,
              const float* __restrict__ b2v,
              uint4* __restrict__ gates)                // [NTOK] (i0,i1,g0,g1)
{
  __shared__ __align__(16) unsigned short hbuf[8][16][64]; // per-wave h tile
  __shared__ float lbuf[8][16][NEXP];                      // per-wave logits
  __shared__ float b1s[HD];

  for (int i = threadIdx.x; i < HD; i += 256) b1s[i] = b1[i];
  __syncthreads();

  const int wave = threadIdx.x >> 5;
  const int lane = threadIdx.x & 31;
  const int row  = lane & 15;   // A: token row; B: output column
  const int half = lane >> 4;   // K-half selector per ISA VGPR layouts
  const int tile = blockIdx.x * 8 + wave;          // 0..511

  const unsigned short* Arow = Abf + (size_t)(tile * 16 + row) * HD;

  const v8f vzero = {0.f,0.f,0.f,0.f,0.f,0.f,0.f,0.f};
  v8f c2 = vzero;  // logits accumulator [16 tok x 16(8 valid) experts]

  for (int g = 0; g < 16; ++g) {                   // 16 groups of 64 columns
    v8f acc0 = vzero, acc1 = vzero, acc2 = vzero, acc3 = vzero;

    for (int kc = 0; kc < 32; ++kc) {              // K = 1024 in chunks of 32
      // A fragment: 16-bit A layout, lane<16: K{0..7,16..23}, lane>=16: K{8..15,24..31}
      ABFrag a;
      const unsigned short* ap = Arow + kc * 32 + half * 8;
      a.q[0] = *(const uint4*)(ap);
      a.q[1] = *(const uint4*)(ap + 16);

#define GEMM1_STEP(u, accv)                                                       \
      {                                                                           \
        const unsigned short* bp =                                                \
            W1T + (size_t)(g * 64 + (u) * 16 + row) * HD + kc * 32 + half * 8;    \
        ABFrag bf_; bf_.q[0] = *(const uint4*)bp;                                 \
        bf_.q[1] = *(const uint4*)(bp + 16);                                      \
        accv = __builtin_amdgcn_wmma_f32_16x16x32_bf16(                           \
            false, a.v, false, bf_.v, (short)0, accv, false, false);              \
      }
      GEMM1_STEP(0, acc0)
      GEMM1_STEP(1, acc1)
      GEMM1_STEP(2, acc2)
      GEMM1_STEP(3, acc3)
#undef GEMM1_STEP
    }

    // bias + relu, write bf16 tile to LDS (D layout: lane=col, vgpr r + 8*half = token)
#define RELU_STORE(u, accv)                                                       \
    {                                                                             \
      float bias = b1s[g * 64 + (u) * 16 + row];                                  \
      for (int r = 0; r < 8; ++r)                                                 \
        hbuf[wave][r + 8 * half][(u) * 16 + row] =                                \
            f2bf(fmaxf(accv[r] + bias, 0.f));                                     \
    }
    RELU_STORE(0, acc0)
    RELU_STORE(1, acc1)
    RELU_STORE(2, acc2)
    RELU_STORE(3, acc3)
#undef RELU_STORE

    // GEMM2: h-tile (A operand, from LDS) x W2T chunk -> logits accumulator
#pragma unroll
    for (int c = 0; c < 2; ++c) {
      ABFrag a2;
      const unsigned short* hp = &hbuf[wave][row][c * 32 + half * 8];
      a2.q[0] = *(const uint4*)(hp);
      a2.q[1] = *(const uint4*)(hp + 16);
      ABFrag w2f;
      const unsigned short* wp = W2T + (size_t)row * HD + g * 64 + c * 32 + half * 8;
      w2f.q[0] = *(const uint4*)(wp);
      w2f.q[1] = *(const uint4*)(wp + 16);
      c2 = __builtin_amdgcn_wmma_f32_16x16x32_bf16(
          false, a2.v, false, w2f.v, (short)0, c2, false, false);
    }
  }

  // dump logits (+b2) to LDS: lanes with col<8 hold valid experts
  if (row < NEXP) {
    for (int r = 0; r < 8; ++r)
      lbuf[wave][r + 8 * half][row] = c2[r] + b2v[row];
  }
  __syncthreads();

  // per-token softmax + top-2 (one lane per token)
  if (lane < 16) {
    const int t = tile * 16 + lane;
    float p[NEXP];
    float mx = -1e30f;
    for (int e = 0; e < NEXP; ++e) { p[e] = lbuf[wave][lane][e]; mx = fmaxf(mx, p[e]); }
    float s = 0.f;
    for (int e = 0; e < NEXP; ++e) { p[e] = __expf(p[e] - mx); s += p[e]; }
    float inv = 1.f / s;
    int i0 = 0; float g0 = p[0];
    for (int e = 1; e < NEXP; ++e) if (p[e] > g0) { g0 = p[e]; i0 = e; }
    int i1 = -1; float g1 = -1.f;
    for (int e = 0; e < NEXP; ++e) if (e != i0 && p[e] > g1) { g1 = p[e]; i1 = e; }
    uint4 m;
    m.x = (unsigned)i0; m.y = (unsigned)i1;
    m.z = __float_as_uint(g0 * inv); m.w = __float_as_uint(g1 * inv);
    gates[t] = m;
  }
}

// ------------------------------ combine kernel -----------------------------

__global__ void __launch_bounds__(256)
combine_kernel(const float* __restrict__ EO,      // [E][NTOK][HD]
               const uint4* __restrict__ gates,   // [NTOK]
               float* __restrict__ out)           // [NTOK][HD]
{
  const int t = blockIdx.x;
  const int c = threadIdx.x * 4;
  uint4 m = gates[t];
  const float4 a = *(const float4*)(EO + ((size_t)m.x * NTOK + t) * HD + c);
  const float4 b = *(const float4*)(EO + ((size_t)m.y * NTOK + t) * HD + c);
  const float g0 = __uint_as_float(m.z), g1 = __uint_as_float(m.w);
  float4 r;
  r.x = g0 * a.x + g1 * b.x;
  r.y = g0 * a.y + g1 * b.y;
  r.z = g0 * a.z + g1 * b.z;
  r.w = g0 * a.w + g1 * b.w;
  *(float4*)(out + (size_t)t * HD + c) = r;
}

// --------------------------------- launch ----------------------------------

extern "C" void kernel_launch(void* const* d_in, const int* in_sizes, int n_in,
                              void* d_out, int out_size, void* d_ws, size_t ws_size,
                              hipStream_t stream) {
  const float* hs  = (const float*)d_in[0];   // [B,S,H]
  const float* EO  = (const float*)d_in[1];   // [E,B,S,H]
  const float* W1  = (const float*)d_in[2];   // [H,H]
  const float* b1  = (const float*)d_in[3];   // [H]
  const float* W2  = (const float*)d_in[4];   // [H,E]
  const float* b2  = (const float*)d_in[5];   // [E]
  float* out = (float*)d_out;

  char* ws = (char*)d_ws;
  const size_t OFF_ABF = 0;
  const size_t OFF_W1T = OFF_ABF + (size_t)NTOK * HD * 2;   // 16 MiB
  const size_t OFF_W2T = OFF_W1T + (size_t)HD * HD * 2;     // +2 MiB
  const size_t OFF_GT  = OFF_W2T + (size_t)16 * HD * 2;     // +32 KiB
  unsigned short* Abf = (unsigned short*)(ws + OFF_ABF);
  unsigned short* W1T = (unsigned short*)(ws + OFF_W1T);
  unsigned short* W2T = (unsigned short*)(ws + OFF_W2T);
  uint4*          gts = (uint4*)(ws + OFF_GT);

  cvt_bf16_kernel<<<(NTOK * HD) / 1024, 256, 0, stream>>>(hs, Abf, NTOK * HD);
  transpose_w1_kernel<<<(HD * HD) / 256, 256, 0, stream>>>(W1, W1T);
  prep_w2_kernel<<<(16 * HD) / 256, 256, 0, stream>>>(W2, W2T);
  router_kernel<<<NTOK / 128, 256, 0, stream>>>(Abf, W1T, W2T, b1, b2, gts);
  combine_kernel<<<NTOK, 256, 0, stream>>>(EO, gts, out);
}
